// SinkhornNormalizer_45827301048807
// MI455X (gfx1250) — compile-verified, matrix operand order checked by reading, not run on hardware
//
#include <hip/hip_runtime.h>
#include <stdint.h>

// Sinkhorn-Knopp, 1024 x (64x64), 100 iterations, fully fused on-chip.
// Linear-domain reformulation: E = exp(x); repeat {E /= colsum; E /= rowsum}; out = E + EPS.
// One workgroup (4 wave32 waves) per matrix. Matrix LDS-resident (stride 68 f32).
// Reductions done as ones-matrix WMMAs (v_wmma_f32_16x16x32_f16, f32 accum) -> the
// C-layout of D replicates the per-column (or per-row) sums, giving free broadcast.
// Input staged via the Tensor Data Mover (tensor_load_to_lds) with pad insertion.

#define SK_EPS   0.001f
#define SK_ITERS 100
#define SK_N     64
#define SK_STR   68          // padded LDS row stride in floats (64 + 4): 16B-aligned rows, de-phased banks
#define SK_BLOCK 128         // 4 waves; one 16-wide slab per wave

typedef __attribute__((ext_vector_type(16))) _Float16     v16h;
typedef __attribute__((ext_vector_type(8)))  float        v8f;
typedef __attribute__((ext_vector_type(4)))  float        f4;
typedef __attribute__((ext_vector_type(4)))  unsigned int v4u;
typedef __attribute__((ext_vector_type(8)))  int          v8i;
typedef __attribute__((ext_vector_type(4)))  int          v4i;

__global__ __launch_bounds__(SK_BLOCK) void SinkhornNormalizer_kernel(const float* __restrict__ x,
                                                                      float* __restrict__ out) {
  __shared__ __align__(16) float Es[SK_N * SK_STR];  // E, stride 68
  __shared__ float rs[SK_N];                         // row-sum broadcast scratch

  const int tid = threadIdx.x;
  const int w   = tid >> 5;          // wave id 0..3
  const int l   = tid & 31;          // lane
  const int hf  = l >> 4;            // lane half (wave32 fragment halves)
  const int q   = l & 15;
  const uint64_t base = (uint64_t)blockIdx.x * (SK_N * SK_N);

  // ---------------- stage input into LDS ----------------
#if __has_builtin(__builtin_amdgcn_tensor_load_to_lds) && __has_builtin(__builtin_amdgcn_s_wait_tensorcnt)
  if (w == 0) {  // one TDM descriptor per workgroup (EXEC-independent, issue from wave 0 only)
    const uint64_t gaddr = (uint64_t)(uintptr_t)(x + base);
    const uint32_t laddr = (uint32_t)(uintptr_t)(&Es[0]);   // low 32 bits = LDS byte offset
    v4u g0; v8i g1; v4i g2; v4i g3; v8i g4;
    g0[0] = 1u;                                             // count=1 (user descriptor)
    g0[1] = laddr;                                          // lds_addr
    g0[2] = (uint32_t)gaddr;                                // global_addr[31:0]
    g0[3] = (uint32_t)((gaddr >> 32) & 0x01FFFFFFu) | (2u << 30);  // global_addr[56:32], type=2
    // group1: data_size=4B(2), pad_enable=1, pad_interval=64 dwords(5), pad_amount=4 dwords(3)
    g1[0] = (int)((2u << 16) | (1u << 20) | (5u << 22) | (3u << 25));
    g1[1] = (int)(64u << 16);   // tensor_dim0 = 64 (bits 79:48)
    g1[2] = (int)(64u << 16);   // tensor_dim1 = 64 (bits 111:80)
    g1[3] = (int)(64u << 16);   // tile_dim0   = 64 (bits 127:112)
    g1[4] = 64;                 // tile_dim1   = 64, tile_dim2 = 0
    g1[5] = 64;                 // tensor_dim0_stride = 64
    g1[6] = 0;  g1[7] = 0;      // tensor_dim1_stride = 0 (2D tile)
    g2[0] = g2[1] = g2[2] = g2[3] = 0;
    g3[0] = g3[1] = g3[2] = g3[3] = 0;
    g4[0] = g4[1] = g4[2] = g4[3] = g4[4] = g4[5] = g4[6] = g4[7] = 0;  // trailing SGPR group (unused)
    __builtin_amdgcn_tensor_load_to_lds(g0, g1, g2, g3, g4, 0);         // 6-arg form (clang-23 toolchain)
    __builtin_amdgcn_s_wait_tensorcnt(0);
  }
  __syncthreads();
  // E = exp(x / TAU), TAU = 1; one transcendental pass total (linear-domain Sinkhorn)
  #pragma unroll
  for (int k = 0; k < 32; ++k) {
    const int j   = tid + k * SK_BLOCK;
    const int idx = (j >> 6) * SK_STR + (j & 63);
    Es[idx] = __expf(Es[idx]);
  }
#else
  #pragma unroll
  for (int k = 0; k < 32; ++k) {
    const int j = tid + k * SK_BLOCK;
    Es[(j >> 6) * SK_STR + (j & 63)] = __expf(x[base + j]);
  }
#endif
  __syncthreads();

  v16h ones;
  #pragma unroll
  for (int i = 0; i < 16; ++i) ones[i] = (_Float16)1.0f;

  #pragma unroll 1
  for (int it = 0; it < SK_ITERS; ++it) {
    // ===== column pass: E[:,c] *= 1/colsum[c]; wave w owns cols [16w,16w+16) =====
    {
      const int c = 16 * w + q;
      float v[32];
      // B fragment here uses contiguous k (16-bit, 32x16): lanes 0-15 hold K=0..15, lanes 16-31 K=16..31.
      // frag0 covers rows 0..31, frag1 rows 32..63 -> every slab element loaded exactly once.
      #pragma unroll
      for (int j = 0; j < 16; ++j) v[j]      = Es[(hf * 16 + j) * SK_STR + c];
      #pragma unroll
      for (int j = 0; j < 16; ++j) v[16 + j] = Es[(32 + hf * 16 + j) * SK_STR + c];
      v16h b0, b1;
      #pragma unroll
      for (int j = 0; j < 16; ++j) { b0[j] = (_Float16)v[j]; b1[j] = (_Float16)v[16 + j]; }
      v8f acc = {};
      acc = __builtin_amdgcn_wmma_f32_16x16x32_f16(false, ones, false, b0, (short)0, acc, false, false);
      acc = __builtin_amdgcn_wmma_f32_16x16x32_f16(false, ones, false, b1, (short)0, acc, false, false);
      // D[i][j] = colsum[16w+j] for every i; lane l holds column 16w+q in every C VGPR.
      const float inv = __builtin_amdgcn_rcpf(acc[0]);  // scale error is self-corrected by next pass
      #pragma unroll
      for (int j = 0; j < 16; ++j) Es[(hf * 16 + j) * SK_STR + c]      = v[j] * inv;
      #pragma unroll
      for (int j = 0; j < 16; ++j) Es[(32 + hf * 16 + j) * SK_STR + c] = v[16 + j] * inv;
    }
    __syncthreads();
    // ===== row pass: E[r,:] *= 1/rowsum[r]; wave w owns rows [16w,16w+16) =====
    {
      const int r = 16 * w + q;
      float* __restrict__ row = &Es[r * SK_STR];
      // A-fragment layout (16-bit, 16x32): lanes 0-15 K=0-7,16-23; lanes 16-31 K=8-15,24-31.
      // -> contiguous 8-float runs: wide (b128) LDS loads. frag1 = cols +32.
      f4 u[8];
      #pragma unroll
      for (int g = 0; g < 4; ++g) {
        u[2 * g]     = *(const f4*)&row[g * 16 + hf * 8];
        u[2 * g + 1] = *(const f4*)&row[g * 16 + hf * 8 + 4];
      }
      v16h a0, a1;
      #pragma unroll
      for (int e = 0; e < 4; ++e) {
        a0[e]      = (_Float16)u[0][e];  a0[4 + e]  = (_Float16)u[1][e];
        a0[8 + e]  = (_Float16)u[2][e];  a0[12 + e] = (_Float16)u[3][e];
        a1[e]      = (_Float16)u[4][e];  a1[4 + e]  = (_Float16)u[5][e];
        a1[8 + e]  = (_Float16)u[6][e];  a1[12 + e] = (_Float16)u[7][e];
      }
      v8f acc = {};
      acc = __builtin_amdgcn_wmma_f32_16x16x32_f16(false, a0, false, ones, (short)0, acc, false, false);
      acc = __builtin_amdgcn_wmma_f32_16x16x32_f16(false, a1, false, ones, (short)0, acc, false, false);
      // acc[m] = rowsum[16w + 8*hf + m]; broadcast via tiny LDS array (in-wave DScnt ordering).
      if (q == 0) {
        #pragma unroll
        for (int m = 0; m < 8; ++m) rs[16 * w + 8 * hf + m] = acc[m];
      }
      const float inv = __builtin_amdgcn_rcpf(rs[r]);
      #pragma unroll
      for (int g = 0; g < 8; ++g) u[g] *= inv;
      #pragma unroll
      for (int g = 0; g < 4; ++g) {
        *(f4*)&row[g * 16 + hf * 8]     = u[2 * g];
        *(f4*)&row[g * 16 + hf * 8 + 4] = u[2 * g + 1];
      }
    }
    __syncthreads();
  }

  // ---------------- out = E + EPS (coalesced b128 stores) ----------------
  #pragma unroll
  for (int k = 0; k < 8; ++k) {
    const int j = tid * 4 + k * 512;
    const int r = j >> 6, c = j & 63;
    f4 val = *(const f4*)&Es[r * SK_STR + c];
    val += SK_EPS;
    *(f4*)&out[base + j] = val;
  }
}

extern "C" void kernel_launch(void* const* d_in, const int* in_sizes, int n_in,
                              void* d_out, int out_size, void* d_ws, size_t ws_size,
                              hipStream_t stream) {
  (void)n_in; (void)d_ws; (void)ws_size; (void)out_size;
  const float* x = (const float*)d_in[0];
  float* out = (float*)d_out;
  const int nmat = in_sizes[0] / (SK_N * SK_N);   // 1024 matrices
  SinkhornNormalizer_kernel<<<nmat, SK_BLOCK, 0, stream>>>(x, out);
}
